// ImprovedAdaptiveDecoder_88210038325432
// MI455X (gfx1250) — matrix operand order repaired and checked by reading, not compiled, not run on hardware
//
#include <hip/hip_runtime.h>
#include <stdint.h>

// ============================================================================
// ImprovedAdaptiveDecoder for MI455X (gfx1250, wave32, WMMA).
//
// All heavy math (attention q/k/v/o, qk^T, attn*v, every MLP linear layer)
// runs through a single WMMA tiled GEMM (f16 in, f32 accumulate) using
// v_wmma_f32_16x16x32_f16.  Global->LDS staging of GEMM tiles uses the CDNA5
// async-copy path (global_load_async_to_lds_b128 + s_wait_asynccnt) with
// ping-pong LDS double buffering so the async engine overlaps the WMMA work.
//
// JAX PRNG streams (fps start index, split angles, repulsion permutation,
// gaussian noise) cannot be reproduced bit-exactly; deterministic hash-based
// substitutes are used (same values every call => deterministic launches).
//
// Workspace requirement: ~2.0 GB (trivial vs 432 GiB HBM).
// ============================================================================

typedef __attribute__((ext_vector_type(16))) _Float16 v16h;
typedef __attribute__((ext_vector_type(8)))  _Float16 v8h;
typedef __attribute__((ext_vector_type(8)))  float    v8f;

#define TPB 256

// ---------------------------------------------------------------------------
// CDNA5 async global->LDS helpers (ISA 15.18.3 opcodes 95..98, ASYNCcnt)
// ---------------------------------------------------------------------------
__device__ __forceinline__ uint32_t lds_off_u32(const void* p) {
  // generic-space LDS pointers carry the LDS byte offset in the low 32 bits
  return (uint32_t)(uintptr_t)p;
}
__device__ __forceinline__ void async_ld_b128(uint32_t ldsOff, const void* g) {
  asm volatile("global_load_async_to_lds_b128 %0, %1, off"
               :: "v"(ldsOff), "v"(g) : "memory");
}
__device__ __forceinline__ void wait_async_all() {
  asm volatile("s_wait_asynccnt 0x0" ::: "memory");
}

// ---------------------------------------------------------------------------
// Deterministic hash RNG (substitute for jax.random)
// ---------------------------------------------------------------------------
__device__ __forceinline__ uint32_t hash_u32(uint32_t x) {
  x ^= x >> 16; x *= 0x7feb352dU;
  x ^= x >> 15; x *= 0x846ca68bU;
  x ^= x >> 16; return x;
}
__device__ __forceinline__ float u01(uint32_t h) {
  return (float)(h >> 8) * (1.0f / 16777216.0f);
}
__device__ __forceinline__ float nrm01(uint32_t h) {
  uint32_t h2 = hash_u32(h ^ 0x85ebca6bU);
  float u1 = fmaxf(u01(h), 1e-7f);
  float u2 = u01(h2);
  return sqrtf(-2.0f * logf(u1)) * cosf(6.2831853071795864f * u2);
}

// ---------------------------------------------------------------------------
// GEMM tile staging into LDS (one 128x32-half A tile + 128x32-half B tile).
//   A is row-major [M x Kp]; LDS layout [m][k], row stride 32 halves.
//   TRANS_B == 1 : B row-major [Np x Kp] -> LDS [n][k] via async copy.
//   TRANS_B == 0 : B row-major [Kp x Np] -> LDS [n][k] transpose via VGPRs.
// ---------------------------------------------------------------------------
template <int TRANS_B>
__device__ __forceinline__ void stage_tile(
    _Float16* dA, _Float16* dB,
    const _Float16* __restrict__ A, const _Float16* __restrict__ Bm,
    int tileM, int tileN, int Np, int Kp, int k0, int t)
{
#pragma unroll
  for (int j = 0; j < 2; ++j) {         // A: 512 x 16B chunks, 2 per thread
    int id  = t + TPB * j;
    int row = id >> 2;
    int seg = id & 3;
    async_ld_b128(lds_off_u32(dA + row * 32 + seg * 8),
                  A + (size_t)(tileM + row) * Kp + k0 + seg * 8);
  }
  if (TRANS_B) {
#pragma unroll
    for (int j = 0; j < 2; ++j) {
      int id  = t + TPB * j;
      int row = id >> 2;
      int seg = id & 3;
      async_ld_b128(lds_off_u32(dB + row * 32 + seg * 8),
                    Bm + (size_t)(tileN + row) * Kp + k0 + seg * 8);
    }
  } else {
#pragma unroll
    for (int j = 0; j < 2; ++j) {
      int id = t + TPB * j;             // 0..511
      int k  = id >> 4;                 // 0..31
      int n0 = (id & 15) * 8;           // 0..120
      v8h bv = *(const v8h*)(Bm + (size_t)(k0 + k) * Np + tileN + n0);
#pragma unroll
      for (int e = 0; e < 8; ++e) dB[(n0 + e) * 32 + k] = bv[e];
    }
  }
}

// ---------------------------------------------------------------------------
// WMMA GEMM: C[M x Np] = alpha * A[M x Kp](f16) * B(f16) + bias + residual
// Requirements: M % 128 == 0, Np % 128 == 0, Kp % 32 == 0 (all buffers padded).
// Block tile 128x128, 8 waves (4x2), wave tile 32x64 (2x4 WMMA accumulators).
// Double-buffered LDS: async copy of tile k+1 overlaps WMMA on tile k.
// ---------------------------------------------------------------------------
template <int TRANS_B>
__global__ __launch_bounds__(TPB) void k_wmma_gemm(
    const _Float16* __restrict__ A, const _Float16* __restrict__ Bm,
    const float* __restrict__ bias, int Nb,
    const float* __restrict__ residual,
    float* __restrict__ C32, _Float16* __restrict__ C16,
    int M, int Np, int Kp, float alpha)
{
  __shared__ _Float16 sA[2][128 * 32];  // [m][k] per buffer
  __shared__ _Float16 sB[2][128 * 32];  // [n][k] per buffer

  const int t     = threadIdx.x;
  const int tileM = blockIdx.y * 128;
  const int tileN = blockIdx.x * 128;
  const int lane  = t & 31;
  const int wid   = t >> 5;
  const int wm    = wid >> 1;           // 0..3  -> 32-row band
  const int wn    = wid & 1;            // 0..1  -> 64-col band
  const int hi    = lane >> 4;          // half-wave select
  const int lr    = lane & 15;

  v8f acc[2][4];
  const v8f vzero = {0.f, 0.f, 0.f, 0.f, 0.f, 0.f, 0.f, 0.f};
#pragma unroll
  for (int f = 0; f < 2; ++f)
#pragma unroll
    for (int g = 0; g < 4; ++g) acc[f][g] = vzero;

  const int nk = Kp >> 5;
  // prologue: stage tile 0 into buffer 0
  stage_tile<TRANS_B>(sA[0], sB[0], A, Bm, tileM, tileN, Np, Kp, 0, t);

  for (int ik = 0; ik < nk; ++ik) {
    const int buf = ik & 1;
    // make buffer `buf` visible to every wave (own asyncs + barrier)
    wait_async_all();
    __syncthreads();

    // kick off the next tile into the other buffer while we compute
    if (ik + 1 < nk) {
      stage_tile<TRANS_B>(sA[buf ^ 1], sB[buf ^ 1], A, Bm,
                          tileM, tileN, Np, Kp, (ik + 1) << 5, t);
      // WGP-scope prefetch of the slab after next (global_prefetch_b8)
      if (ik + 2 < nk)
        __builtin_prefetch(A + (size_t)(tileM + (t >> 1)) * Kp + ((ik + 2) << 5),
                           0, 3);
    }

    // ---- fragments: contiguous k order (k = 16*hi + e), same for A and B
    const _Float16* cA = sA[buf];
    const _Float16* cB = sB[buf];
    v16h af[2];
#pragma unroll
    for (int f = 0; f < 2; ++f) {
      // A 16x32: lane m=lr, halves e -> k = 16*hi + e (contiguous 32B)
      const _Float16* base = cA + (wm * 32 + f * 16 + lr) * 32;
      af[f] = *(const v16h*)(base + 16 * hi);
    }
    v16h bf[4];
#pragma unroll
    for (int g = 0; g < 4; ++g) {
      // B 32x16: lane n=lr, halves e -> k = 16*hi + e (contiguous 32B)
      const _Float16* base = cB + (wn * 64 + g * 16 + lr) * 32;
      bf[g] = *(const v16h*)(base + 16 * hi);
    }
#pragma unroll
    for (int f = 0; f < 2; ++f)
#pragma unroll
      for (int g = 0; g < 4; ++g)
        acc[f][g] = __builtin_amdgcn_wmma_f32_16x16x32_f16(
            false, af[f], false, bf[g], (short)0, acc[f][g], false, false);
  }

  // ---- epilogue: C/D layout -> element (m = r + 8*hi, n = lr) per 16x16 tile
#pragma unroll
  for (int f = 0; f < 2; ++f)
    for (int g = 0; g < 4; ++g)
#pragma unroll
      for (int r = 0; r < 8; ++r) {
        int m = tileM + wm * 32 + f * 16 + hi * 8 + r;
        int n = tileN + wn * 64 + g * 16 + lr;
        float v = alpha * acc[f][g][r];
        if (bias && n < Nb) v += bias[n];
        size_t o = (size_t)m * Np + n;
        if (residual) v += residual[o];
        if (C32) C32[o] = v;
        if (C16) C16[o] = (_Float16)v;
      }
}

// ---------------------------------------------------------------------------
// LayerNorm (+ReLU) over C columns of one row; writes f32 and/or f16 outputs
// ---------------------------------------------------------------------------
__global__ __launch_bounds__(TPB) void k_ln_relu(
    const float* __restrict__ X, const float* __restrict__ gam,
    const float* __restrict__ bet, float* __restrict__ o32,
    _Float16* __restrict__ o16, int C)
{
  __shared__ float red[TPB];
  size_t row = blockIdx.x;
  int t = threadIdx.x;
  const float* x = X + row * C;

  float s = 0.f;
  for (int c = t; c < C; c += TPB) s += x[c];
  red[t] = s; __syncthreads();
  for (int k = TPB / 2; k > 0; k >>= 1) { if (t < k) red[t] += red[t + k]; __syncthreads(); }
  float mean = red[0] / C; __syncthreads();

  float v = 0.f;
  for (int c = t; c < C; c += TPB) { float d = x[c] - mean; v += d * d; }
  red[t] = v; __syncthreads();
  for (int k = TPB / 2; k > 0; k >>= 1) { if (t < k) red[t] += red[t + k]; __syncthreads(); }
  float inv = rsqrtf(red[0] / C + 1e-5f); __syncthreads();

  for (int c = t; c < C; c += TPB) {
    float y = (x[c] - mean) * inv * gam[c] + bet[c];
    y = fmaxf(y, 0.f);
    if (o32) o32[row * C + c] = y;
    if (o16) o16[row * C + c] = (_Float16)y;
  }
}

// ---------------------------------------------------------------------------
// Row softmax (f32 logits -> f16 probs), C = 2048
// ---------------------------------------------------------------------------
__global__ __launch_bounds__(TPB) void k_softmax(
    const float* __restrict__ L, _Float16* __restrict__ P, int C)
{
  __shared__ float red[TPB];
  size_t row = blockIdx.x;
  int t = threadIdx.x;
  const float* x = L + row * C;

  float m = -1e30f;
  for (int c = t; c < C; c += TPB) m = fmaxf(m, x[c]);
  red[t] = m; __syncthreads();
  for (int k = TPB / 2; k > 0; k >>= 1) { if (t < k) red[t] = fmaxf(red[t], red[t + k]); __syncthreads(); }
  m = red[0]; __syncthreads();

  float s = 0.f;
  for (int c = t; c < C; c += TPB) s += __expf(x[c] - m);
  red[t] = s; __syncthreads();
  for (int k = TPB / 2; k > 0; k >>= 1) { if (t < k) red[t] += red[t + k]; __syncthreads(); }
  float inv = 1.f / red[0];

  for (int c = t; c < C; c += TPB) P[row * C + c] = (_Float16)(__expf(x[c] - m) * inv);
}

// ---------------------------------------------------------------------------
// Farthest point sampling: N=2048, S=64 seeds per batch (start index 0)
// ---------------------------------------------------------------------------
__global__ __launch_bounds__(TPB) void k_fps(
    const float* __restrict__ xyz, int* __restrict__ seeds, int N, int S)
{
  __shared__ float dist[2048];
  __shared__ float rv[TPB];
  __shared__ int   ri[TPB];
  int b = blockIdx.x, t = threadIdx.x;
  const float* X = xyz + (size_t)b * N * 3;
  for (int i = t; i < N; i += TPB) dist[i] = 1e10f;
  int cur = 0;
  __syncthreads();
  for (int it = 0; it < S; ++it) {
    if (t == 0) seeds[b * S + it] = cur;
    float cx = X[cur * 3 + 0], cy = X[cur * 3 + 1], cz = X[cur * 3 + 2];
    float bm = -1.f; int bi = 0;
    for (int i = t; i < N; i += TPB) {
      float dx = X[i * 3] - cx, dy = X[i * 3 + 1] - cy, dz = X[i * 3 + 2] - cz;
      float nd = fminf(dist[i], dx * dx + dy * dy + dz * dz);
      dist[i] = nd;
      if (nd > bm) { bm = nd; bi = i; }
    }
    rv[t] = bm; ri[t] = bi; __syncthreads();
    for (int k = TPB / 2; k > 0; k >>= 1) {
      if (t < k && rv[t + k] > rv[t]) { rv[t] = rv[t + k]; ri[t] = ri[t + k]; }
      __syncthreads();
    }
    cur = ri[0];
    __syncthreads();
  }
}

// --------------------------- elementwise kernels ---------------------------
__global__ void k_f32_to_f16(const float* s, _Float16* d, long long n) {
  size_t i = (size_t)blockIdx.x * TPB + threadIdx.x;
  if (i < (size_t)n) d[i] = (_Float16)s[i];
}
__global__ void k_copy_f32(const float* s, float* d, long long n) {
  size_t i = (size_t)blockIdx.x * TPB + threadIdx.x;
  if (i < (size_t)n) d[i] = s[i];
}
__global__ void k_pack_w(const float* w, _Float16* dst, int K, int N, int Np,
                         long long total) {
  size_t i = (size_t)blockIdx.x * TPB + threadIdx.x;
  if (i >= (size_t)total) return;
  int n = (int)(i % Np);
  int k = (int)(i / Np);
  dst[i] = (k < K && n < N) ? (_Float16)w[(size_t)k * N + n] : (_Float16)0.f;
}
__global__ void k_gather_rows(const float* src, const int* idx, float* out,
                              int rowsPerB, int srcRows, int C, long long total) {
  size_t i = (size_t)blockIdx.x * TPB + threadIdx.x;
  if (i >= (size_t)total) return;
  size_t r = i / C;
  int c = (int)(i % C);
  int b = (int)(r / rowsPerB);
  out[i] = src[((size_t)b * srcRows + idx[r]) * C + c];
}
__constant__ float c_gs[4] = {-0.1f, -0.0333333333f, 0.0333333333f, 0.1f};
__global__ void k_pack_fold(const float* sfeat, const float* sxyz,
                            _Float16* dst, long long total) {
  size_t i = (size_t)blockIdx.x * TPB + threadIdx.x;
  if (i >= (size_t)total) return;
  int c = (int)(i % 544);
  size_t r = i / 544;     // row = ((b*64+s)*16 + p)
  size_t sr = r / 16;
  int p = (int)(r % 16);
  float v = 0.f;
  if (c < 512)       v = sfeat[sr * 512 + c];
  else if (c == 512) v = c_gs[p >> 2];
  else if (c == 513) v = c_gs[p & 3];
  else if (c < 517)  v = sxyz[sr * 3 + (c - 514)];
  dst[i] = (_Float16)v;
}
__global__ void k_add_pos(const float* mlpOut, const float* sxyz, float* xyz,
                          long long total) {
  size_t i = (size_t)blockIdx.x * TPB + threadIdx.x;
  if (i >= (size_t)total) return;
  size_t r = i / 3;
  int c = (int)(i % 3);
  xyz[i] = mlpOut[r * 128 + c] + sxyz[(r / 16) * 3 + c];
}
__global__ void k_mean_local(const float* locf, float* mean) {
  int b = blockIdx.x, t = threadIdx.x;
  for (int c = t; c < 512; c += TPB) {
    float s = 0.f;
    for (int k = 0; k < 256; ++k) s += locf[((size_t)b * 256 + k) * 512 + c];
    mean[b * 512 + c] = s * (1.f / 256.f);
  }
}
__global__ void k_broadcast_feat(const float* mean, float* feat, int N,
                                 long long total) {
  size_t i = (size_t)blockIdx.x * TPB + threadIdx.x;
  if (i >= (size_t)total) return;
  size_t r = i / 512;
  int c = (int)(i % 512);
  int b = (int)(r / N);
  feat[i] = mean[b * 512 + c];
}
__global__ __launch_bounds__(TPB) void k_std3(const float* xyz, int N, float* out) {
  __shared__ float red[TPB];
  int b = blockIdx.x, t = threadIdx.x;
  const float* X = xyz + (size_t)b * N * 3;
  for (int c = 0; c < 3; ++c) {
    float s = 0.f, ss = 0.f;
    for (int i = t; i < N; i += TPB) { float v = X[i * 3 + c]; s += v; ss += v * v; }
    red[t] = s; __syncthreads();
    for (int k = TPB / 2; k > 0; k >>= 1) { if (t < k) red[t] += red[t + k]; __syncthreads(); }
    float sum = red[0]; __syncthreads();
    red[t] = ss; __syncthreads();
    for (int k = TPB / 2; k > 0; k >>= 1) { if (t < k) red[t] += red[t + k]; __syncthreads(); }
    float sqs = red[0]; __syncthreads();
    if (t == 0) {
      float var = (sqs - sum * sum / N) / (float)(N - 1);
      out[b * 3 + c] = sqrtf(fmaxf(var, 0.f));
    }
    __syncthreads();
  }
}
__global__ __launch_bounds__(TPB) void k_nearest(const float* xyz, const float* locf,
                                                 int* outIdx, int N) {
  __shared__ float lp[256 * 3];
  int b = blockIdx.y, t = threadIdx.x;
  const float* L = locf + (size_t)b * 256 * 512;
  for (int j = t; j < 256; j += TPB) {
    lp[j * 3 + 0] = L[(size_t)j * 512 + 0];
    lp[j * 3 + 1] = L[(size_t)j * 512 + 1];
    lp[j * 3 + 2] = L[(size_t)j * 512 + 2];
  }
  __syncthreads();
  int p = blockIdx.x * TPB + t;
  if (p >= N) return;
  const float* X = xyz + ((size_t)b * N + p) * 3;
  float x = X[0], y = X[1], z = X[2];
  float best = 1e30f; int bi = 0;
  for (int j = 0; j < 256; ++j) {
    float dx = x - lp[j * 3], dy = y - lp[j * 3 + 1], dz = z - lp[j * 3 + 2];
    float d = dx * dx + dy * dy + dz * dz;
    if (d < best) { best = d; bi = j; }
  }
  outIdx[(size_t)b * N + p] = bi;
}
__global__ void k_pack_combined(const float* feat, const float* locf, const int* nidx,
                               const float* xyz, const float* spread,
                               _Float16* dst, int N, long long total) {
  size_t i = (size_t)blockIdx.x * TPB + threadIdx.x;
  if (i >= (size_t)total) return;
  int c = (int)(i % 1056);
  size_t r = i / 1056;
  int b = (int)(r / N);
  float v = 0.f;
  if (c < 512)       v = feat[r * 512 + c];
  else if (c < 1024) v = locf[((size_t)b * 256 + nidx[r]) * 512 + (c - 512)];
  else if (c < 1027) v = xyz[r * 3 + (c - 1024)];
  else if (c < 1030) v = spread[b * 3 + (c - 1027)];
  dst[i] = (_Float16)v;
}
__global__ void k_pack_split(const float* featR, const float* locf, const int* nidx,
                             const float* xyz, _Float16* dst, int N, long long total) {
  size_t i = (size_t)blockIdx.x * TPB + threadIdx.x;
  if (i >= (size_t)total) return;
  int c = (int)(i % 1056);
  size_t r = i / 1056;
  int b = (int)(r / N);
  float v = 0.f;
  if (c < 512)       v = featR[r * 512 + c];
  else if (c < 1024) v = locf[((size_t)b * 256 + nidx[r]) * 512 + (c - 512)];
  else if (c < 1027) v = xyz[r * 3 + (c - 1024)];
  dst[i] = (_Float16)v;
}
__global__ void k_pack_refine(const float* feat, const float* xyz, const float* spread,
                              _Float16* dst, int N2, long long total) {
  size_t i = (size_t)blockIdx.x * TPB + threadIdx.x;
  if (i >= (size_t)total) return;
  int c = (int)(i % 544);
  size_t r = i / 544;
  int b = (int)(r / N2);
  float v = 0.f;
  if (c < 512)      v = feat[r * 512 + c];
  else if (c < 515) v = xyz[r * 3 + (c - 512)];
  else if (c < 518) v = spread[b * 3 + (c - 515)];
  dst[i] = (_Float16)v;
}
__global__ void k_split_apply(const float* sp, const float* xin, float* xout,
                              int N, unsigned seed, long long R) {
  size_t r = (size_t)blockIdx.x * TPB + threadIdx.x;
  if (r >= (size_t)R) return;
  int b = (int)(r / N), n = (int)(r % N);
  const float* s = sp + r * 128;
  float x0 = xin[r * 3], x1 = xin[r * 3 + 1], x2 = xin[r * 3 + 2];
  float ang = u01(hash_u32(seed * 2654435761u ^ (unsigned)r)) * 6.2831853071795864f;
  float o20 = s[3] * 0.05f + cosf(ang) * 0.02f;
  float o21 = s[4] * 0.05f + sinf(ang) * 0.02f;
  float o22 = s[5] * 0.05f;
  size_t r1 = (size_t)b * 2 * N + n;
  size_t r2 = r1 + N;
  xout[r1 * 3 + 0] = x0 + s[0] * 0.05f;
  xout[r1 * 3 + 1] = x1 + s[1] * 0.05f;
  xout[r1 * 3 + 2] = x2 + s[2] * 0.05f;
  xout[r2 * 3 + 0] = x0 + o20;
  xout[r2 * 3 + 1] = x1 + o21;
  xout[r2 * 3 + 2] = x2 + o22;
}
__global__ void k_dup_feat(const float* featR, float* out, int N, long long total) {
  size_t i = (size_t)blockIdx.x * TPB + threadIdx.x;
  if (i >= (size_t)total) return;
  size_t r = i / 512;
  int c = (int)(i % 512);
  int b = (int)(r / (2 * N));
  int j = (int)(r % (2 * N));
  if (j >= N) j -= N;
  out[i] = featR[((size_t)b * N + j) * 512 + c];
}
__global__ void k_apply_offset(float* xyz, const float* mlpOut, const float* scales,
                               int stage, long long total) {
  size_t i = (size_t)blockIdx.x * TPB + threadIdx.x;
  if (i >= (size_t)total) return;
  size_t r = i / 3;
  int c = (int)(i % 3);
  xyz[i] += mlpOut[r * 128 + c] * fabsf(scales[stage]);
}
__global__ __launch_bounds__(TPB) void k_repulsion(float* xyz, int N2, float min_dist) {
  __shared__ float xs[1000 * 3];
  int b = blockIdx.y, t = threadIdx.x;
  float* X = xyz + (size_t)b * N2 * 3;
  for (int j = t; j < 1000; j += TPB) {
    xs[j * 3 + 0] = X[j * 3 + 0];
    xs[j * 3 + 1] = X[j * 3 + 1];
    xs[j * 3 + 2] = X[j * 3 + 2];
  }
  __syncthreads();
  int i = blockIdx.x * TPB + t;
  if (i >= 1000) return;
  float x = xs[i * 3], y = xs[i * 3 + 1], z = xs[i * 3 + 2];
  float fx = 0.f, fy = 0.f, fz = 0.f;
  for (int j = 0; j < 1000; ++j) {
    if (j == i) continue;
    float dx = x - xs[j * 3], dy = y - xs[j * 3 + 1], dz = z - xs[j * 3 + 2];
    float d = sqrtf(fmaxf(dx * dx + dy * dy + dz * dz, 1e-12f));
    if (d < min_dist) {
      float inv = 1.f / (d + 1e-8f);
      fx += dx * inv; fy += dy * inv; fz += dz * inv;
    }
  }
  X[i * 3 + 0] = x + fx * 0.01f;
  X[i * 3 + 1] = y + fy * 0.01f;
  X[i * 3 + 2] = z + fz * 0.01f;
}
__global__ void k_noise_scale(const float* spread, float* out) {
  if (threadIdx.x == 0 && blockIdx.x == 0) {
    float s = 0.f;
    for (int i = 0; i < 24; ++i) s += spread[i];
    out[0] = fmaxf(0.2f - s * (1.f / 24.f), 0.f);
  }
}
__global__ void k_add_noise(float* xyz, const float* scale, unsigned seed,
                            long long total) {
  size_t i = (size_t)blockIdx.x * TPB + threadIdx.x;
  if (i >= (size_t)total) return;
  xyz[i] += nrm01(hash_u32(seed * 0x9E3779B9u ^ (unsigned)i)) * scale[0];
}

// ===========================================================================
// Host orchestration
// ===========================================================================
// d_in pytree flatten order (sorted dict keys, None pruned):
//   0 partial_xyz  1 partial_feat  2 local_features
//   fold:   blocks[i]{b,beta,g,w} at 3+4i (i=0..2); final{b,w} at 15,16
//   offset_scales: 17
//   props[p] (p=0..4) base 18+8p: k.b,k.w,o.b,o.w,q.b,q.w,v.b,v.w
//   refiners[i] base 58+10i: blk0{b,beta,g,w}, blk1{...}, final{b=+8,w=+9}
//   stages[i] base 98+18i: refine blk0(+0..3) blk1(+4..7);
//                          split blk0(+8..11) blk1(+12..15) final{b=+16,w=+17}
extern "C" void kernel_launch(void* const* d_in, const int* in_sizes, int n_in,
                              void* d_out, int out_size, void* d_ws, size_t ws_size,
                              hipStream_t stream) {
  (void)in_sizes; (void)n_in; (void)out_size; (void)ws_size;

  const int Bn = 8, Mrows = 2048;
  const size_t RMAX = (size_t)Bn * 16384;

  auto P = [&](int i) -> const float* { return (const float*)d_in[i]; };
  const float* pxyz  = P(0);
  const float* pfeat = P(1);
  const float* locf  = P(2);

  // bump allocator over workspace
  char* wsb = (char*)d_ws;
  size_t cur = 0;
  auto alloc = [&](size_t bytes) -> void* {
    void* p = wsb + cur;
    cur = (cur + bytes + 255) & ~(size_t)255;
    return p;
  };
  auto nblk = [](size_t n) { return dim3((unsigned)((n + TPB - 1) / TPB)); };

  _Float16* H1    = (_Float16*)alloc(RMAX * 1056 * sizeof(_Float16));
  _Float16* H2    = (_Float16*)alloc(RMAX * 512 * sizeof(_Float16));
  _Float16* qH    = (_Float16*)alloc(RMAX * 512 * sizeof(_Float16));
  _Float16* attnH = (_Float16*)alloc(RMAX * 512 * sizeof(_Float16));
  _Float16* pfH   = (_Float16*)alloc((size_t)Bn * Mrows * 512 * sizeof(_Float16));
  _Float16* kH    = (_Float16*)alloc((size_t)Bn * Mrows * 512 * sizeof(_Float16));
  _Float16* vH    = (_Float16*)alloc((size_t)Bn * Mrows * 512 * sizeof(_Float16));
  _Float16* probs = (_Float16*)alloc((size_t)16384 * 2048 * sizeof(_Float16));
  float* logits = (float*)alloc((size_t)16384 * 2048 * sizeof(float));
  float* T1     = (float*)alloc(RMAX * 512 * sizeof(float));
  float* featA  = (float*)alloc(RMAX * 512 * sizeof(float));
  float* featB  = (float*)alloc(RMAX * 512 * sizeof(float));
  float* featR  = (float*)alloc((RMAX / 2) * 512 * sizeof(float));
  float* xyzA   = (float*)alloc(RMAX * 3 * sizeof(float));
  float* xyzB   = (float*)alloc(RMAX * 3 * sizeof(float));
  float* spreadB = (float*)alloc(256);
  float* scaleB  = (float*)alloc(256);
  float* meanL   = (float*)alloc((size_t)Bn * 512 * sizeof(float));
  float* seedXyz = (float*)alloc((size_t)Bn * 64 * 3 * sizeof(float));
  float* seedFeat = (float*)alloc((size_t)Bn * 64 * 512 * sizeof(float));
  int* seedIdx = (int*)alloc((size_t)Bn * 64 * sizeof(int));
  int* nearI   = (int*)alloc((RMAX / 2) * sizeof(int));

  auto gemm = [&](int transB, const _Float16* Aop, const _Float16* Bop,
                  const float* bias, int Nb, const float* resid,
                  float* C32, _Float16* C16, int Mr, int Np, int Kp, float alpha) {
    dim3 g(Np / 128, Mr / 128);
    if (transB)
      k_wmma_gemm<1><<<g, TPB, 0, stream>>>(Aop, Bop, bias, Nb, resid, C32, C16, Mr, Np, Kp, alpha);
    else
      k_wmma_gemm<0><<<g, TPB, 0, stream>>>(Aop, Bop, bias, Nb, resid, C32, C16, Mr, Np, Kp, alpha);
  };
  auto packW = [&](int wI, int K, int Nreal, int Np) -> const _Float16* {
    int Kp = (K + 31) & ~31;
    _Float16* dst = (_Float16*)alloc((size_t)Kp * Np * sizeof(_Float16));
    long long tot = (long long)Kp * Np;
    k_pack_w<<<nblk((size_t)tot), TPB, 0, stream>>>(P(wI), dst, K, Nreal, Np, tot);
    return dst;
  };
  auto layerLN = [&](const _Float16* in16, int Mr, int K, int N,
                     int wI, int bI, int gI, int betaI,
                     float* o32, _Float16* o16) {
    int Kp = (K + 31) & ~31;
    const _Float16* W = packW(wI, K, N, N);
    gemm(0, in16, W, P(bI), N, nullptr, T1, nullptr, Mr, N, Kp, 1.f);
    k_ln_relu<<<Mr, TPB, 0, stream>>>(T1, P(gI), P(betaI), o32, o16, N);
  };
  auto runProp = [&](int p, const float* fin, float* fout, int N) {
    int R = Bn * N;
    int base = 18 + 8 * p;  // k.b,k.w,o.b,o.w,q.b,q.w,v.b,v.w
    k_f32_to_f16<<<nblk((size_t)R * 512), TPB, 0, stream>>>(fin, H2, (long long)R * 512);
    const _Float16* Wq = packW(base + 5, 512, 512, 512);
    gemm(0, H2, Wq, P(base + 4), 512, nullptr, nullptr, qH, R, 512, 512, 1.f);
    const _Float16* Wk = packW(base + 1, 512, 512, 512);
    gemm(0, pfH, Wk, P(base + 0), 512, nullptr, nullptr, kH, Bn * Mrows, 512, 512, 1.f);
    const _Float16* Wv = packW(base + 7, 512, 512, 512);
    gemm(0, pfH, Wv, P(base + 6), 512, nullptr, nullptr, vH, Bn * Mrows, 512, 512, 1.f);
    for (int b = 0; b < Bn; ++b) {
      gemm(1, qH + (size_t)b * N * 512, kH + (size_t)b * Mrows * 512,
           nullptr, 0, nullptr, logits, nullptr, N, Mrows, 512, 0.04419417382f);
      k_softmax<<<N, TPB, 0, stream>>>(logits, probs, Mrows);
      gemm(0, probs, vH + (size_t)b * Mrows * 512, nullptr, 0, nullptr,
           nullptr, attnH + (size_t)b * N * 512, N, 512, Mrows, 1.f);
    }
    const _Float16* Wo = packW(base + 3, 512, 512, 512);
    gemm(0, attnH, Wo, P(base + 2), 512, fin, fout, nullptr, R, 512, 512, 1.f);
  };

  // ---- once: partial_feat -> f16
  k_f32_to_f16<<<nblk((size_t)Bn * Mrows * 512), TPB, 0, stream>>>(
      pfeat, pfH, (long long)Bn * Mrows * 512);

  // ---- FPS seeds + gathers
  k_fps<<<Bn, TPB, 0, stream>>>(pxyz, seedIdx, Mrows, 64);
  k_gather_rows<<<nblk((size_t)Bn * 64 * 3), TPB, 0, stream>>>(
      pxyz, seedIdx, seedXyz, 64, Mrows, 3, (long long)Bn * 64 * 3);
  k_gather_rows<<<nblk((size_t)Bn * 64 * 512), TPB, 0, stream>>>(
      pfeat, seedIdx, seedFeat, 64, Mrows, 512, (long long)Bn * 64 * 512);

  // ---- folding MLP: 517 -> 512 -> 256 -> 128 -> 3 (+pos)
  k_pack_fold<<<nblk((size_t)8192 * 544), TPB, 0, stream>>>(
      seedFeat, seedXyz, H1, (long long)8192 * 544);
  layerLN(H1, 8192, 517, 512,  6,  3,  5,  4, nullptr, H2);
  layerLN(H2, 8192, 512, 256, 10,  7,  9,  8, nullptr, H2);
  layerLN(H2, 8192, 256, 128, 14, 11, 13, 12, nullptr, H2);
  { const _Float16* W = packW(16, 128, 3, 128);
    gemm(0, H2, W, P(15), 3, nullptr, T1, nullptr, 8192, 128, 128, 1.f); }
  k_add_pos<<<nblk((size_t)8192 * 3), TPB, 0, stream>>>(T1, seedXyz, xyzA,
                                                        (long long)8192 * 3);

  // ---- feat init: mean(local_features) broadcast; attention prop 0 @ N=1024
  k_mean_local<<<Bn, TPB, 0, stream>>>(locf, meanL);
  k_broadcast_feat<<<nblk((size_t)Bn * 1024 * 512), TPB, 0, stream>>>(
      meanL, featA, 1024, (long long)Bn * 1024 * 512);
  runProp(0, featA, featB, 1024);

  float* featCur = featB;
  float* xyzCur = xyzA;
  float* xyzNext = xyzB;

  for (int i = 0; i < 4; ++i) {
    int N = 1024 << i;
    int R = Bn * N;
    int s  = 98 + 18 * i;   // stage param base
    int rf = 58 + 10 * i;   // refiner param base

    k_std3<<<Bn, TPB, 0, stream>>>(xyzCur, N, spreadB);
    k_nearest<<<dim3((N + TPB - 1) / TPB, Bn), TPB, 0, stream>>>(xyzCur, locf, nearI, N);

    // refine MLP: 1030 -> 512 -> 512 (feat_r)
    k_pack_combined<<<nblk((size_t)R * 1056), TPB, 0, stream>>>(
        featCur, locf, nearI, xyzCur, spreadB, H1, N, (long long)R * 1056);
    layerLN(H1, R, 1030, 512, s + 3, s + 0, s + 2, s + 1, nullptr, H2);
    layerLN(H2, R,  512, 512, s + 7, s + 4, s + 6, s + 5, featR, nullptr);

    // split MLP: 1027 -> 512 -> 128 -> 6
    k_pack_split<<<nblk((size_t)R * 1056), TPB, 0, stream>>>(
        featR, locf, nearI, xyzCur, H1, N, (long long)R * 1056);
    layerLN(H1, R, 1027, 512, s + 11, s + 8,  s + 10, s + 9,  nullptr, H2);
    layerLN(H2, R,  512, 128, s + 15, s + 12, s + 14, s + 13, nullptr, H2);
    { const _Float16* W = packW(s + 17, 128, 6, 128);
      gemm(0, H2, W, P(s + 16), 6, nullptr, T1, nullptr, R, 128, 128, 1.f); }
    k_split_apply<<<nblk((size_t)R), TPB, 0, stream>>>(T1, xyzCur, xyzNext, N,
                                                       1000u + i, (long long)R);
    k_dup_feat<<<nblk((size_t)Bn * 2 * N * 512), TPB, 0, stream>>>(
        featR, featA, N, (long long)Bn * 2 * N * 512);

    // attention prop i+1 @ 2N
    runProp(i + 1, featA, featB, 2 * N);
    featCur = featB;

    // refiner MLP: 518 -> 512 -> 256 -> 3, scaled offset
    k_std3<<<Bn, TPB, 0, stream>>>(xyzNext, 2 * N, spreadB);
    k_pack_refine<<<nblk((size_t)2 * R * 544), TPB, 0, stream>>>(
        featB, xyzNext, spreadB, H1, 2 * N, (long long)2 * R * 544);
    layerLN(H1, 2 * R, 518, 512, rf + 3, rf + 0, rf + 2, rf + 1, nullptr, H2);
    layerLN(H2, 2 * R, 512, 256, rf + 7, rf + 4, rf + 6, rf + 5, nullptr, H2);
    { const _Float16* W = packW(rf + 9, 256, 3, 128);
      gemm(0, H2, W, P(rf + 8), 3, nullptr, T1, nullptr, 2 * R, 128, 256, 1.f); }
    k_apply_offset<<<nblk((size_t)2 * R * 3), TPB, 0, stream>>>(
        xyzNext, T1, P(17), i, (long long)2 * R * 3);

    // repulsion on first 1000 points (deterministic stand-in for random perm)
    k_repulsion<<<dim3(4, Bn), TPB, 0, stream>>>(xyzNext, 2 * N, 0.01f / (float)(1 << i));

    // spread-floor noise
    k_std3<<<Bn, TPB, 0, stream>>>(xyzNext, 2 * N, spreadB);
    k_noise_scale<<<1, 32, 0, stream>>>(spreadB, scaleB);
    k_add_noise<<<nblk((size_t)2 * R * 3), TPB, 0, stream>>>(
        xyzNext, scaleB, 2000u + i, (long long)2 * R * 3);

    float* tmp = xyzCur; xyzCur = xyzNext; xyzNext = tmp;
  }

  k_copy_f32<<<nblk((size_t)Bn * 16384 * 3), TPB, 0, stream>>>(
      xyzCur, (float*)d_out, (long long)Bn * 16384 * 3);
}
